// DeepSeekV3MoEGEMM_28037546508849
// MI455X (gfx1250) — matrix operand order repaired and checked by reading, not compiled, not run on hardware
//
#include <hip/hip_runtime.h>

#define T_TOK 4096
#define H_DIM 1024
#define I_DIM 512
#define E_NUM 16
#define M_TILE 32
#define XPAD 1032   // 1024 + 8 (keeps 16B alignment: 1032*2 % 16 == 0)
#define HPAD 520    // 512 + 8  (520*2 % 16 == 0)

typedef __attribute__((ext_vector_type(16))) __bf16 bf16x16;
typedef __attribute__((ext_vector_type(8)))  __bf16 bf16x8;
typedef __attribute__((ext_vector_type(8)))  float  f32x8;

union BFrag { bf16x16 v; bf16x8 h[2]; };

__device__ __forceinline__ unsigned short f2bfu(float f) {
  union { float f; unsigned u; } c; c.f = f;
  unsigned r = c.u + 0x7FFFu + ((c.u >> 16) & 1u);   // round-to-nearest-even
  return (unsigned short)(r >> 16);
}

// ---------------- fp32 -> bf16 weight conversion (one HBM pass) --------------
__global__ void cvt_bf16_k(const float* __restrict__ src,
                           unsigned short* __restrict__ dst, int n4) {
  int stride = gridDim.x * blockDim.x;
  for (int i = blockIdx.x * blockDim.x + threadIdx.x; i < n4; i += stride) {
    float4 v = reinterpret_cast<const float4*>(src)[i];
    ushort4 p;
    p.x = f2bfu(v.x); p.y = f2bfu(v.y); p.z = f2bfu(v.z); p.w = f2bfu(v.w);
    reinterpret_cast<ushort4*>(dst)[i] = p;
  }
}

// ---------------- router: sigmoid(x @ Wr^T) ---------------------------------
__global__ void router_logits_k(const float* __restrict__ x,
                                const float* __restrict__ rw,
                                float* __restrict__ scores) {
  int g = blockIdx.x * blockDim.x + threadIdx.x;   // T*E threads
  int t = g >> 4, e = g & 15;
  const float4* xr = reinterpret_cast<const float4*>(x + (size_t)t * H_DIM);
  const float4* wr = reinterpret_cast<const float4*>(rw + (size_t)e * H_DIM);
  float acc = 0.f;
  for (int i = 0; i < H_DIM / 4; ++i) {
    float4 a = xr[i], b = wr[i];
    acc += a.x * b.x + a.y * b.y + a.z * b.z + a.w * b.w;
  }
  scores[g] = 1.f / (1.f + __expf(-acc));
}

// ---------------- router: group-limited top-k -> dense combine[T][E] --------
__global__ void router_topk_k(const float* __restrict__ scores,
                              const float* __restrict__ bias,
                              float* __restrict__ combine) {
  int t = blockIdx.x * blockDim.x + threadIdx.x;
  if (t >= T_TOK) return;
  float s[E_NUM], sc[E_NUM];
#pragma unroll
  for (int e = 0; e < E_NUM; ++e) { s[e] = scores[t * E_NUM + e]; sc[e] = s[e] + bias[e]; }
  float gs[4];
#pragma unroll
  for (int g = 0; g < 4; ++g) {            // sum of top-2 within each group of 4
    float m1 = -1e30f, m2 = -1e30f;
#pragma unroll
    for (int k = 0; k < 4; ++k) {
      float v = sc[g * 4 + k];
      if (v > m1) { m2 = m1; m1 = v; } else if (v > m2) { m2 = v; }
    }
    gs[g] = m1 + m2;
  }
  int g1 = 0;
#pragma unroll
  for (int g = 1; g < 4; ++g) if (gs[g] > gs[g1]) g1 = g;   // strict > : lower idx wins ties
  int g2 = (g1 == 0) ? 1 : 0;
#pragma unroll
  for (int g = 0; g < 4; ++g) if (g != g1 && gs[g] > gs[g2]) g2 = g;
  unsigned gmask = (1u << g1) | (1u << g2);
  float msk[E_NUM];
#pragma unroll
  for (int e = 0; e < E_NUM; ++e)
    msk[e] = ((gmask >> (e >> 2)) & 1u) ? sc[e] : -1e9f;
  unsigned chosen = 0;
  int idx[4]; float w[4]; float wsum = 0.f;
  for (int k = 0; k < 4; ++k) {
    int best = 0; float bv = -1e30f;
#pragma unroll
    for (int e = 0; e < E_NUM; ++e)
      if (!((chosen >> e) & 1u) && msk[e] > bv) { bv = msk[e]; best = e; }
    chosen |= 1u << best;
    idx[k] = best; w[k] = s[best]; wsum += s[best];
  }
  float inv = 2.5f / (wsum + 1e-20f);
  float c[E_NUM];
#pragma unroll
  for (int e = 0; e < E_NUM; ++e) c[e] = 0.f;
  for (int k = 0; k < 4; ++k) c[idx[k]] = w[k] * inv;
#pragma unroll
  for (int e = 0; e < E_NUM; ++e) combine[(size_t)t * E_NUM + e] = c[e];
}

// ---------------- fused SwiGLU expert GEMMs (bf16 WMMA, f32 acc) ------------
// Each B (weight) fragment is loaded from L2 exactly once and applied to both
// M-row-tiles -> per-WG weight traffic hits the 48MB minimum (32 flop/L2-byte).
__global__ void __launch_bounds__(256)
moe_expert_k(const float* __restrict__ x, const float* __restrict__ combine,
             const unsigned short* __restrict__ wgb,
             const unsigned short* __restrict__ wub,
             const unsigned short* __restrict__ wdb,
             float* __restrict__ out) {
  __shared__ unsigned short sX[M_TILE][XPAD];   // x tile, bf16          (~66 KB)
  __shared__ unsigned short sH[M_TILE][HPAD];   // silu(g)*u*combine     (~33 KB)
  __shared__ float sCmb[M_TILE][E_NUM];

  const int tid    = threadIdx.x;
  const int wave   = tid >> 5;
  const int lane   = tid & 31;
  const int lane15 = lane & 15;
  const int khalf  = lane >> 4;
  const int t0     = blockIdx.x * M_TILE;

  // load + convert X tile (tile rows are contiguous tokens)
  const float4* xt = reinterpret_cast<const float4*>(x + (size_t)t0 * H_DIM);
  for (int i = tid; i < M_TILE * H_DIM / 4; i += 256) {
    float4 v = xt[i];
    int elem = i * 4;
    int row = elem >> 10;
    int col = elem & (H_DIM - 1);
    sX[row][col + 0] = f2bfu(v.x);
    sX[row][col + 1] = f2bfu(v.y);
    sX[row][col + 2] = f2bfu(v.z);
    sX[row][col + 3] = f2bfu(v.w);
  }
  for (int i = tid; i < M_TILE * E_NUM; i += 256)
    sCmb[i >> 4][i & 15] = combine[(size_t)(t0 + (i >> 4)) * E_NUM + (i & 15)];
  __syncthreads();

  const f32x8 vzero = {0.f, 0.f, 0.f, 0.f, 0.f, 0.f, 0.f, 0.f};
  f32x8 acc[16];   // persistent out tile: acc[j]=mi0, acc[8+j]=mi1, cn=wave*8+j
#pragma unroll
  for (int j = 0; j < 16; ++j) acc[j] = vzero;

  for (int e = 0; e < E_NUM; ++e) {
    // ---- phase 1: G = X Wg^T, U = X Wu^T, h = silu(G)*U*combine ----
#pragma unroll 1
    for (int nn = 0; nn < 4; ++nn) {
      const int ni = wave * 4 + nn;
      const size_t wbase = (size_t)e * I_DIM * H_DIM +
                           (size_t)(ni * 16 + lane15) * H_DIM + (size_t)(khalf * 8);
      const unsigned short* grow = wgb + wbase;
      const unsigned short* urow = wub + wbase;
      f32x8 g0 = vzero, u0 = vzero, g1 = vzero, u1 = vzero;
#pragma unroll 4
      for (int k0 = 0; k0 < H_DIM; k0 += 32) {
        BFrag a0, a1, bg, bu;
        a0.h[0] = *reinterpret_cast<const bf16x8*>(&sX[lane15][k0 + khalf * 8]);
        a0.h[1] = *reinterpret_cast<const bf16x8*>(&sX[lane15][k0 + 16 + khalf * 8]);
        a1.h[0] = *reinterpret_cast<const bf16x8*>(&sX[16 + lane15][k0 + khalf * 8]);
        a1.h[1] = *reinterpret_cast<const bf16x8*>(&sX[16 + lane15][k0 + 16 + khalf * 8]);
        bg.h[0] = *reinterpret_cast<const bf16x8*>(grow + k0);
        bg.h[1] = *reinterpret_cast<const bf16x8*>(grow + k0 + 16);
        bu.h[0] = *reinterpret_cast<const bf16x8*>(urow + k0);
        bu.h[1] = *reinterpret_cast<const bf16x8*>(urow + k0 + 16);
        g0 = __builtin_amdgcn_wmma_f32_16x16x32_bf16(false, a0.v, false, bg.v,
                                                     (short)0, g0, false, false);
        u0 = __builtin_amdgcn_wmma_f32_16x16x32_bf16(false, a0.v, false, bu.v,
                                                     (short)0, u0, false, false);
        g1 = __builtin_amdgcn_wmma_f32_16x16x32_bf16(false, a1.v, false, bg.v,
                                                     (short)0, g1, false, false);
        u1 = __builtin_amdgcn_wmma_f32_16x16x32_bf16(false, a1.v, false, bu.v,
                                                     (short)0, u1, false, false);
      }
      const int ncol = ni * 16 + lane15;
#pragma unroll
      for (int r = 0; r < 8; ++r) {
        const int m0 = khalf * 8 + r;
        const int m1 = 16 + m0;
        float gv0 = g0[r], uv0 = u0[r];
        float gv1 = g1[r], uv1 = u1[r];
        // silu via v_rcp_f32 (avoids IEEE div expansion)
        float h0 = gv0 * __builtin_amdgcn_rcpf(1.f + __expf(-gv0)) * uv0 * sCmb[m0][e];
        float h1 = gv1 * __builtin_amdgcn_rcpf(1.f + __expf(-gv1)) * uv1 * sCmb[m1][e];
        sH[m0][ncol] = f2bfu(h0);
        sH[m1][ncol] = f2bfu(h1);
      }
    }
    __syncthreads();
    // ---- phase 2: out += h Wd^T (combine already folded into h) ----
#pragma unroll 1
    for (int j = 0; j < 8; ++j) {
      const int cn = wave * 8 + j;
      const unsigned short* drow = wdb + (size_t)e * H_DIM * I_DIM +
                                   (size_t)(cn * 16 + lane15) * I_DIM +
                                   (size_t)(khalf * 8);
      f32x8 c0 = acc[j], c1 = acc[8 + j];
#pragma unroll 4
      for (int k0 = 0; k0 < I_DIM; k0 += 32) {
        BFrag a0, a1, b;
        a0.h[0] = *reinterpret_cast<const bf16x8*>(&sH[lane15][k0 + khalf * 8]);
        a0.h[1] = *reinterpret_cast<const bf16x8*>(&sH[lane15][k0 + 16 + khalf * 8]);
        a1.h[0] = *reinterpret_cast<const bf16x8*>(&sH[16 + lane15][k0 + khalf * 8]);
        a1.h[1] = *reinterpret_cast<const bf16x8*>(&sH[16 + lane15][k0 + 16 + khalf * 8]);
        b.h[0] = *reinterpret_cast<const bf16x8*>(drow + k0);
        b.h[1] = *reinterpret_cast<const bf16x8*>(drow + k0 + 16);
        c0 = __builtin_amdgcn_wmma_f32_16x16x32_bf16(false, a0.v, false, b.v,
                                                     (short)0, c0, false, false);
        c1 = __builtin_amdgcn_wmma_f32_16x16x32_bf16(false, a1.v, false, b.v,
                                                     (short)0, c1, false, false);
      }
      acc[j] = c0; acc[8 + j] = c1;
    }
    __syncthreads();
  }

  // write out (C/D layout: col = lane&15, rows = khalf*8 + r)
#pragma unroll
  for (int j = 0; j < 16; ++j) {
    const int col = (wave * 8 + (j & 7)) * 16 + lane15;
    const int rbase = (j >> 3) * 16 + khalf * 8;
#pragma unroll
    for (int r = 0; r < 8; ++r)
      out[(size_t)(t0 + rbase + r) * H_DIM + col] = acc[j][r];
  }
}

// ---------------------------------------------------------------------------
extern "C" void kernel_launch(void* const* d_in, const int* in_sizes, int n_in,
                              void* d_out, int out_size, void* d_ws, size_t ws_size,
                              hipStream_t stream) {
  (void)in_sizes; (void)n_in; (void)out_size; (void)ws_size;
  const float* x  = (const float*)d_in[0];
  const float* rw = (const float*)d_in[1];
  const float* rb = (const float*)d_in[2];
  const float* wg = (const float*)d_in[3];
  const float* wu = (const float*)d_in[4];
  const float* wd = (const float*)d_in[5];
  float* out = (float*)d_out;

  char* ws = (char*)d_ws;
  float* scores  = (float*)(ws);                       // T*E*4   = 256 KB
  float* combine = (float*)(ws + (size_t)262144);      // T*E*4   = 256 KB
  unsigned short* wgb = (unsigned short*)(ws + (size_t)524288);         // 16 MB
  unsigned short* wub = wgb + (size_t)E_NUM * I_DIM * H_DIM;            // 16 MB
  unsigned short* wdb = wub + (size_t)E_NUM * I_DIM * H_DIM;            // 16 MB

  const int n4 = E_NUM * I_DIM * H_DIM / 4;            // 2,097,152 float4s each
  cvt_bf16_k<<<4096, 256, 0, stream>>>(wg, wgb, n4);
  cvt_bf16_k<<<4096, 256, 0, stream>>>(wu, wub, n4);
  cvt_bf16_k<<<4096, 256, 0, stream>>>(wd, wdb, n4);

  router_logits_k<<<T_TOK * E_NUM / 256, 256, 0, stream>>>(x, rw, scores);
  router_topk_k<<<T_TOK / 256, 256, 0, stream>>>(scores, rb, combine);

  moe_expert_k<<<T_TOK / M_TILE, 256, 0, stream>>>(x, combine, wgb, wub, wdb, out);
}
